// GeneralGNN_22170621182103
// MI455X (gfx1250) — matrix operand order repaired and checked by reading, not compiled, not run on hardware
//
#include <hip/hip_runtime.h>
#include <hip/hip_bf16.h>

typedef _Float16 h16;
typedef __attribute__((ext_vector_type(16))) _Float16 v16h;
typedef __attribute__((ext_vector_type(8)))  _Float16 v8h;
typedef __attribute__((ext_vector_type(8)))  float    v8f;

#define LDS_FENCE() asm volatile("s_wait_dscnt 0" ::: "memory")

__device__ __forceinline__ float gelu_f(float x) {
    return 0.5f * x * (1.0f + erff(x * 0.7071067811865476f));
}

// A-matrix 16x32 f16 layout: lane = (row | kg<<4); row=lane&15 holds
// k in [kg*8, kg*8+8) and [16+kg*8, 16+kg*8+8)  (two 16B runs).
__device__ __forceinline__ v16h ld_a(const h16* rowp, int kg) {
    v8h lo = *(const v8h*)(rowp + kg * 8);
    v8h hi = *(const v8h*)(rowp + 16 + kg * 8);
    v16h a;
#pragma unroll
    for (int i = 0; i < 8; ++i) { a[i] = lo[i]; a[i + 8] = hi[i]; }
    return a;
}

// B-matrix 32x16 f16: lane col = lane&15, holds k in [ko + kg*8, +8) and [ko + 16 + kg*8, +8).
// Weights stored (out,in) row-major -> per-lane two 16B runs.
__device__ __forceinline__ v16h ld_b(const h16* W, int Kin, int col, int ko, int kg) {
    const h16* p = W + (size_t)col * Kin + ko + kg * 8;
    v8h lo = *(const v8h*)p;
    v8h hi = *(const v8h*)(p + 16);
    v16h b;
#pragma unroll
    for (int i = 0; i < 8; ++i) { b[i] = lo[i]; b[i + 8] = hi[i]; }
    return b;
}

__device__ __forceinline__ v8f wmma16(v16h a, v16h b, v8f c) {
    return __builtin_amdgcn_wmma_f32_16x16x32_f16(false, a, false, b, (short)0, c, false, false);
}

__device__ __forceinline__ void atomicMaxF(float* addr, float val) {
    unsigned int* ua = (unsigned int*)addr;
    unsigned int old = *ua;
    while (true) {
        float f = __uint_as_float(old);
        if (f >= val) break;
        unsigned int assumed = old;
        old = atomicCAS(ua, assumed, __float_as_uint(val));
        if (old == assumed) break;
    }
}

// ---------------- elementwise / reduction helpers ----------------

__global__ void k_f2h(const float* __restrict__ x, h16* __restrict__ y, int n) {
    int i = blockIdx.x * 256 + threadIdx.x;
    if (i < n) y[i] = (h16)x[i];
}

__global__ void k_padwb3(const float* __restrict__ Wb3, const float* __restrict__ bb3,
                         h16* __restrict__ Wb3p, float* __restrict__ bb3p) {
    int i = blockIdx.x * 256 + threadIdx.x;
    if (i < 16 * 128) Wb3p[i] = (i < 4 * 128) ? (h16)Wb3[i] : (h16)0.0f;
    if (i < 16) bb3p[i] = (i < 4) ? bb3[i] : 0.0f;
}

__global__ void k_init(float* __restrict__ m4, float* __restrict__ s4,
                       float* __restrict__ agg, int N) {
    int i = blockIdx.x * 256 + threadIdx.x;
    if (i < N * 128) agg[i] = 0.0f;
    if (i < N * 4) { m4[i] = -3.0e38f; s4[i] = 0.0f; }
}

__global__ void k_segmax(const float* __restrict__ w4, const int* __restrict__ eidx,
                         float* __restrict__ m4, int E) {
    int i = blockIdx.x * 256 + threadIdx.x;
    if (i >= E * 4) return;
    int e = i >> 2, h = i & 3;
    int n = eidx[e];
    atomicMaxF(&m4[n * 4 + h], w4[i]);
}

__global__ void k_segsum(const float* __restrict__ w4, const float* __restrict__ m4,
                         const int* __restrict__ eidx, float* __restrict__ s4, int E) {
    int i = blockIdx.x * 256 + threadIdx.x;
    if (i >= E * 4) return;
    int e = i >> 2, h = i & 3;
    int n = eidx[e];
    atomicAdd(&s4[n * 4 + h], expf(w4[i] - m4[n * 4 + h]));
}

__global__ void k_segagg(const float* __restrict__ w4, const float* __restrict__ m4,
                         const int* __restrict__ eidx, const h16* __restrict__ V16,
                         float* __restrict__ agg, long long EL) {
    long long i = (long long)blockIdx.x * 256 + threadIdx.x;
    if (i >= EL * 128) return;
    int e = (int)(i >> 7), d = (int)(i & 127), h = d >> 5;
    int n = eidx[e];
    float ex = expf(w4[e * 4 + h] - m4[n * 4 + h]);
    atomicAdd(&agg[(size_t)n * 128 + d], ex * (float)V16[i]);
}

// per-feature batchnorm stats: 128 blocks, 256 threads each
__global__ void k_stats(const float* __restrict__ x, int rows,
                        float* __restrict__ mean, float* __restrict__ inv) {
    int j = blockIdx.x;
    __shared__ float s1[256], s2[256];
    float a = 0.f, b = 0.f;
    for (int i = threadIdx.x; i < rows; i += 256) {
        float v = x[(size_t)i * 128 + j];
        a += v; b += v * v;
    }
    s1[threadIdx.x] = a; s2[threadIdx.x] = b;
    __syncthreads();
    for (int off = 128; off > 0; off >>= 1) {
        if (threadIdx.x < off) { s1[threadIdx.x] += s1[threadIdx.x + off]; s2[threadIdx.x] += s2[threadIdx.x + off]; }
        __syncthreads();
    }
    if (threadIdx.x == 0) {
        float mu = s1[0] / rows;
        float var = s2[0] / rows - mu * mu;
        mean[j] = mu;
        inv[j] = rsqrtf(var + 1e-5f);
    }
}

__global__ void k_norm(const float* __restrict__ x, const float* __restrict__ mean,
                       const float* __restrict__ inv, const float* __restrict__ g,
                       const float* __restrict__ b, float* __restrict__ y32,
                       h16* __restrict__ y16, long long n) {
    long long i = (long long)blockIdx.x * 256 + threadIdx.x;
    if (i >= n) return;
    int j = (int)(i & 127);
    float v = (x[i] - mean[j]) * inv[j] * g[j] + b[j];
    if (y32) y32[i] = v;
    if (y16) y16[i] = (h16)v;
}

// ---------------- WMMA edge kernels ----------------

// NeighborAttention edge part: V = MLP3([hE|hV[dst]]), w = MLP3([hV[src]|hE|hV[dst]]) / sqrt(32)
__global__ __launch_bounds__(128) void k_edge_attn(
    const h16* __restrict__ hE16, const h16* __restrict__ hV16,
    const int* __restrict__ eidx,
    const h16* __restrict__ Wv1, const float* __restrict__ bv1,
    const h16* __restrict__ Wv2, const float* __restrict__ bv2,
    const h16* __restrict__ Wv3, const float* __restrict__ bv3,
    const h16* __restrict__ Wb1, const float* __restrict__ bb1,
    const h16* __restrict__ Wb2, const float* __restrict__ bb2,
    const h16* __restrict__ Wb3p, const float* __restrict__ bb3p,
    h16* __restrict__ V16, float* __restrict__ wout, int E, int ntiles)
{
    __shared__ __align__(16) h16 lds[4 * 2 * 2048];
    int wave = threadIdx.x >> 5, lane = threadIdx.x & 31;
    int tile = blockIdx.x * 4 + wave;
    if (tile >= ntiles) return;
    int c = lane & 15, kg = lane >> 4;
    int e0 = tile * 16;
    int er = e0 + c; if (er >= E) er = E - 1;
    int si = eidx[er], di = eidx[E + er];
    h16* buf0 = lds + wave * 4096;
    h16* buf1 = buf0 + 2048;
    const h16* rowE = hE16 + (size_t)er * 128;
    const h16* rowS = hV16 + (size_t)si * 128;
    const h16* rowD = hV16 + (size_t)di * 128;

    // ---- V-MLP layer 1: K=256 ([hE | hV[dst]]) ----
    {
        v16h a[8];
#pragma unroll
        for (int kk = 0; kk < 8; ++kk) {
            int ko = kk * 32;
            a[kk] = (ko < 128) ? ld_a(rowE + ko, kg) : ld_a(rowD + (ko - 128), kg);
        }
#pragma unroll
        for (int no = 0; no < 128; no += 16) {
            v8f acc = {};
#pragma unroll
            for (int kk = 0; kk < 8; ++kk) acc = wmma16(a[kk], ld_b(Wv1, 256, no + c, kk * 32, kg), acc);
            float bb = bv1[no + c];
#pragma unroll
            for (int rr = 0; rr < 8; ++rr) buf0[(rr + 8 * kg) * 128 + no + c] = (h16)gelu_f(acc[rr] + bb);
        }
    }
    LDS_FENCE();
    // ---- V-MLP layer 2: K=128 ----
    {
        v16h a[4];
#pragma unroll
        for (int kk = 0; kk < 4; ++kk) a[kk] = ld_a(buf0 + c * 128 + kk * 32, kg);
#pragma unroll
        for (int no = 0; no < 128; no += 16) {
            v8f acc = {};
#pragma unroll
            for (int kk = 0; kk < 4; ++kk) acc = wmma16(a[kk], ld_b(Wv2, 128, no + c, kk * 32, kg), acc);
            float bb = bv2[no + c];
#pragma unroll
            for (int rr = 0; rr < 8; ++rr) buf1[(rr + 8 * kg) * 128 + no + c] = (h16)gelu_f(acc[rr] + bb);
        }
    }
    LDS_FENCE();
    // ---- V-MLP layer 3: K=128, no act, store V fp16 ----
    {
        v16h a[4];
#pragma unroll
        for (int kk = 0; kk < 4; ++kk) a[kk] = ld_a(buf1 + c * 128 + kk * 32, kg);
#pragma unroll
        for (int no = 0; no < 128; no += 16) {
            v8f acc = {};
#pragma unroll
            for (int kk = 0; kk < 4; ++kk) acc = wmma16(a[kk], ld_b(Wv3, 128, no + c, kk * 32, kg), acc);
            float bb = bv3[no + c];
#pragma unroll
            for (int rr = 0; rr < 8; ++rr) {
                int ee = e0 + rr + 8 * kg;
                if (ee < E) V16[(size_t)ee * 128 + no + c] = (h16)(acc[rr] + bb);
            }
        }
    }
    LDS_FENCE();
    // ---- Bias MLP layer 1: K=384 ([hV[src] | hE | hV[dst]]) ----
    {
        v16h a[12];
#pragma unroll
        for (int kk = 0; kk < 12; ++kk) {
            int ko = kk * 32;
            const h16* p = (ko < 128) ? (rowS + ko) : (ko < 256) ? (rowE + ko - 128) : (rowD + ko - 256);
            a[kk] = ld_a(p, kg);
        }
#pragma unroll
        for (int no = 0; no < 128; no += 16) {
            v8f acc = {};
#pragma unroll
            for (int kk = 0; kk < 12; ++kk) acc = wmma16(a[kk], ld_b(Wb1, 384, no + c, kk * 32, kg), acc);
            float bb = bb1[no + c];
#pragma unroll
            for (int rr = 0; rr < 8; ++rr) buf0[(rr + 8 * kg) * 128 + no + c] = (h16)gelu_f(acc[rr] + bb);
        }
    }
    LDS_FENCE();
    // ---- Bias MLP layer 2: K=128 ----
    {
        v16h a[4];
#pragma unroll
        for (int kk = 0; kk < 4; ++kk) a[kk] = ld_a(buf0 + c * 128 + kk * 32, kg);
#pragma unroll
        for (int no = 0; no < 128; no += 16) {
            v8f acc = {};
#pragma unroll
            for (int kk = 0; kk < 4; ++kk) acc = wmma16(a[kk], ld_b(Wb2, 128, no + c, kk * 32, kg), acc);
            float bb = bb2[no + c];
#pragma unroll
            for (int rr = 0; rr < 8; ++rr) buf1[(rr + 8 * kg) * 128 + no + c] = (h16)gelu_f(acc[rr] + bb);
        }
    }
    LDS_FENCE();
    // ---- Bias MLP layer 3: K=128, N=4 (zero-padded to 16), scale 1/sqrt(32) ----
    {
        v16h a[4];
#pragma unroll
        for (int kk = 0; kk < 4; ++kk) a[kk] = ld_a(buf1 + c * 128 + kk * 32, kg);
        v8f acc = {};
#pragma unroll
        for (int kk = 0; kk < 4; ++kk) acc = wmma16(a[kk], ld_b(Wb3p, 128, c, kk * 32, kg), acc);
        if (c < 4) {
            float bb = bb3p[c];
            const float scale = 0.17677669529663687f; // 1/sqrt(32)
#pragma unroll
            for (int rr = 0; rr < 8; ++rr) {
                int ee = e0 + rr + 8 * kg;
                if (ee < E) wout[(size_t)ee * 4 + c] = (acc[rr] + bb) * scale;
            }
        }
    }
}

// EdgeMLP: msg = MLP3([hV2[src]|hE|hV2[dst]]); xe = hE(fp32) + msg  (pre-norm)
__global__ __launch_bounds__(128) void k_edge_mlp(
    const h16* __restrict__ hV216, const h16* __restrict__ hE16,
    const float* __restrict__ hE32, const int* __restrict__ eidx,
    const h16* __restrict__ We1, const float* __restrict__ be1,
    const h16* __restrict__ We2, const float* __restrict__ be2,
    const h16* __restrict__ We3, const float* __restrict__ be3,
    float* __restrict__ xe, int E, int ntiles)
{
    __shared__ __align__(16) h16 lds[4 * 2 * 2048];
    int wave = threadIdx.x >> 5, lane = threadIdx.x & 31;
    int tile = blockIdx.x * 4 + wave;
    if (tile >= ntiles) return;
    int c = lane & 15, kg = lane >> 4;
    int e0 = tile * 16;
    int er = e0 + c; if (er >= E) er = E - 1;
    int si = eidx[er], di = eidx[E + er];
    h16* buf0 = lds + wave * 4096;
    h16* buf1 = buf0 + 2048;
    const h16* rowE = hE16 + (size_t)er * 128;
    const h16* rowS = hV216 + (size_t)si * 128;
    const h16* rowD = hV216 + (size_t)di * 128;

    {
        v16h a[12];
#pragma unroll
        for (int kk = 0; kk < 12; ++kk) {
            int ko = kk * 32;
            const h16* p = (ko < 128) ? (rowS + ko) : (ko < 256) ? (rowE + ko - 128) : (rowD + ko - 256);
            a[kk] = ld_a(p, kg);
        }
#pragma unroll
        for (int no = 0; no < 128; no += 16) {
            v8f acc = {};
#pragma unroll
            for (int kk = 0; kk < 12; ++kk) acc = wmma16(a[kk], ld_b(We1, 384, no + c, kk * 32, kg), acc);
            float bb = be1[no + c];
#pragma unroll
            for (int rr = 0; rr < 8; ++rr) buf0[(rr + 8 * kg) * 128 + no + c] = (h16)gelu_f(acc[rr] + bb);
        }
    }
    LDS_FENCE();
    {
        v16h a[4];
#pragma unroll
        for (int kk = 0; kk < 4; ++kk) a[kk] = ld_a(buf0 + c * 128 + kk * 32, kg);
#pragma unroll
        for (int no = 0; no < 128; no += 16) {
            v8f acc = {};
#pragma unroll
            for (int kk = 0; kk < 4; ++kk) acc = wmma16(a[kk], ld_b(We2, 128, no + c, kk * 32, kg), acc);
            float bb = be2[no + c];
#pragma unroll
            for (int rr = 0; rr < 8; ++rr) buf1[(rr + 8 * kg) * 128 + no + c] = (h16)gelu_f(acc[rr] + bb);
        }
    }
    LDS_FENCE();
    {
        v16h a[4];
#pragma unroll
        for (int kk = 0; kk < 4; ++kk) a[kk] = ld_a(buf1 + c * 128 + kk * 32, kg);
#pragma unroll
        for (int no = 0; no < 128; no += 16) {
            v8f acc = {};
#pragma unroll
            for (int kk = 0; kk < 4; ++kk) acc = wmma16(a[kk], ld_b(We3, 128, no + c, kk * 32, kg), acc);
            float bb = be3[no + c];
#pragma unroll
            for (int rr = 0; rr < 8; ++rr) {
                int ee = e0 + rr + 8 * kg;
                if (ee < E) xe[(size_t)ee * 128 + no + c] = hE32[(size_t)ee * 128 + no + c] + acc[rr] + bb;
            }
        }
    }
}

// ---------------- WMMA node kernels ----------------

// dh = (agg/s) @ Wo^T ; x0 = hV + dh
__global__ __launch_bounds__(128) void k_node_wo(
    const float* __restrict__ agg, const float* __restrict__ s4,
    const h16* __restrict__ Wo16, const float* __restrict__ hV32,
    float* __restrict__ x0, int N, int ntiles)
{
    int wave = threadIdx.x >> 5, lane = threadIdx.x & 31;
    int tile = blockIdx.x * 4 + wave;
    if (tile >= ntiles) return;
    int c = lane & 15, kg = lane >> 4;
    int row = tile * 16 + c;
    int rA = (row < N) ? row : (N - 1);
    const float* rp = agg + (size_t)rA * 128;
    float inv[4];
#pragma unroll
    for (int h = 0; h < 4; ++h) {
        float sv = s4[rA * 4 + h];
        inv[h] = (sv > 0.f) ? (1.f / sv) : 0.f;
    }
    v16h a[4];
#pragma unroll
    for (int kk = 0; kk < 4; ++kk) {
        int ko = kk * 32;
        v16h t;
#pragma unroll
        for (int i = 0; i < 8; ++i) {
            int k0 = ko + kg * 8 + i;
            int k1 = ko + 16 + kg * 8 + i;
            t[i]     = (h16)(rp[k0] * inv[k0 >> 5]);
            t[i + 8] = (h16)(rp[k1] * inv[k1 >> 5]);
        }
        a[kk] = t;
    }
#pragma unroll
    for (int no = 0; no < 128; no += 16) {
        v8f acc = {};
#pragma unroll
        for (int kk = 0; kk < 4; ++kk) acc = wmma16(a[kk], ld_b(Wo16, 128, no + c, kk * 32, kg), acc);
#pragma unroll
        for (int rr = 0; rr < 8; ++rr) {
            int r2 = tile * 16 + rr + 8 * kg;
            if (r2 < N) x0[(size_t)r2 * 128 + no + c] = hV32[(size_t)r2 * 128 + no + c] + acc[rr];
        }
    }
}

// t = gelu(hV1 @ Wd1^T + bd1), 128 -> 512
__global__ __launch_bounds__(128) void k_ffn1(
    const h16* __restrict__ hV1_16, const h16* __restrict__ Wd116,
    const float* __restrict__ bd1, h16* __restrict__ t16, int N, int ntiles)
{
    int wave = threadIdx.x >> 5, lane = threadIdx.x & 31;
    int tile = blockIdx.x * 4 + wave;
    if (tile >= ntiles) return;
    int c = lane & 15, kg = lane >> 4;
    int row = tile * 16 + c;
    int rA = (row < N) ? row : (N - 1);
    const h16* rowp = hV1_16 + (size_t)rA * 128;
    v16h a[4];
#pragma unroll
    for (int kk = 0; kk < 4; ++kk) a[kk] = ld_a(rowp + kk * 32, kg);
    for (int no = 0; no < 512; no += 16) {
        v8f acc = {};
#pragma unroll
        for (int kk = 0; kk < 4; ++kk) acc = wmma16(a[kk], ld_b(Wd116, 128, no + c, kk * 32, kg), acc);
        float bb = bd1[no + c];
#pragma unroll
        for (int rr = 0; rr < 8; ++rr) {
            int r2 = tile * 16 + rr + 8 * kg;
            if (r2 < N) t16[(size_t)r2 * 512 + no + c] = (h16)gelu_f(acc[rr] + bb);
        }
    }
}

// dh = t @ Wd2^T + bd2 (512 -> 128); x1 = hV1 + dh
__global__ __launch_bounds__(128) void k_ffn2(
    const h16* __restrict__ t16, const h16* __restrict__ Wd216,
    const float* __restrict__ bd2, const float* __restrict__ hV1_32,
    float* __restrict__ x1, int N, int ntiles)
{
    int wave = threadIdx.x >> 5, lane = threadIdx.x & 31;
    int tile = blockIdx.x * 4 + wave;
    if (tile >= ntiles) return;
    int c = lane & 15, kg = lane >> 4;
    int row = tile * 16 + c;
    int rA = (row < N) ? row : (N - 1);
    const h16* rowp = t16 + (size_t)rA * 512;
    v16h a[16];
#pragma unroll
    for (int kk = 0; kk < 16; ++kk) a[kk] = ld_a(rowp + kk * 32, kg);
#pragma unroll
    for (int no = 0; no < 128; no += 16) {
        v8f acc = {};
#pragma unroll
        for (int kk = 0; kk < 16; ++kk) acc = wmma16(a[kk], ld_b(Wd216, 512, no + c, kk * 32, kg), acc);
        float bb = bd2[no + c];
#pragma unroll
        for (int rr = 0; rr < 8; ++rr) {
            int r2 = tile * 16 + rr + 8 * kg;
            if (r2 < N) x1[(size_t)r2 * 128 + no + c] = hV1_32[(size_t)r2 * 128 + no + c] + acc[rr] + bb;
        }
    }
}

// ---------------- host ----------------

extern "C" void kernel_launch(void* const* d_in, const int* in_sizes, int n_in,
                              void* d_out, int out_size, void* d_ws, size_t ws_size,
                              hipStream_t stream) {
    const int N = in_sizes[0] / 128;
    const int E = in_sizes[1] / 128;

    const float* hV32 = (const float*)d_in[0];
    const float* hE32 = (const float*)d_in[1];
    const int*   eidx = (const int*)d_in[2];
    // d_in[3] = batch_id (unused: context is a no-op)
    const float* Wv1 = (const float*)d_in[4];  const float* bv1 = (const float*)d_in[5];
    const float* Wv2 = (const float*)d_in[6];  const float* bv2 = (const float*)d_in[7];
    const float* Wv3 = (const float*)d_in[8];  const float* bv3 = (const float*)d_in[9];
    const float* Wb1 = (const float*)d_in[10]; const float* bb1 = (const float*)d_in[11];
    const float* Wb2 = (const float*)d_in[12]; const float* bb2 = (const float*)d_in[13];
    const float* Wb3 = (const float*)d_in[14]; const float* bb3 = (const float*)d_in[15];
    const float* Wo  = (const float*)d_in[16];
    const float* g0  = (const float*)d_in[17]; const float* b0  = (const float*)d_in[18];
    const float* g1  = (const float*)d_in[19]; const float* b1  = (const float*)d_in[20];
    const float* Wd1 = (const float*)d_in[21]; const float* bd1 = (const float*)d_in[22];
    const float* Wd2 = (const float*)d_in[23]; const float* bd2 = (const float*)d_in[24];
    const float* We1 = (const float*)d_in[25]; const float* be1 = (const float*)d_in[26];
    const float* We2 = (const float*)d_in[27]; const float* be2 = (const float*)d_in[28];
    const float* We3 = (const float*)d_in[29]; const float* be3 = (const float*)d_in[30];
    const float* ge  = (const float*)d_in[31]; const float* bEn = (const float*)d_in[32];

    float* out_hV = (float*)d_out;
    float* out_hE = (float*)d_out + (size_t)N * 128;

    // ---- workspace bump allocator (256B aligned) ----
    size_t off = 0;
    char* base = (char*)d_ws;
    auto alloc = [&](size_t bytes) -> char* {
        size_t a = (off + 255) & ~(size_t)255;
        off = a + bytes;
        return base + a;
    };
    h16* hV16    = (h16*)alloc((size_t)N * 128 * 2);
    h16* hE16    = (h16*)alloc((size_t)E * 128 * 2);
    h16* V16     = (h16*)alloc((size_t)E * 128 * 2);
    h16* hV1_16  = (h16*)alloc((size_t)N * 128 * 2);
    h16* t16     = (h16*)alloc((size_t)N * 512 * 2);
    h16* hV2_16  = (h16*)alloc((size_t)N * 128 * 2);
    h16* Wv116 = (h16*)alloc(128 * 256 * 2);
    h16* Wv216 = (h16*)alloc(128 * 128 * 2);
    h16* Wv316 = (h16*)alloc(128 * 128 * 2);
    h16* Wb116 = (h16*)alloc(128 * 384 * 2);
    h16* Wb216 = (h16*)alloc(128 * 128 * 2);
    h16* Wb3p  = (h16*)alloc(16 * 128 * 2);
    h16* Wo16  = (h16*)alloc(128 * 128 * 2);
    h16* Wd116 = (h16*)alloc(512 * 128 * 2);
    h16* Wd216 = (h16*)alloc(128 * 512 * 2);
    h16* We116 = (h16*)alloc(128 * 384 * 2);
    h16* We216 = (h16*)alloc(128 * 128 * 2);
    h16* We316 = (h16*)alloc(128 * 128 * 2);
    float* wE   = (float*)alloc((size_t)E * 4 * 4);
    float* m4   = (float*)alloc((size_t)N * 4 * 4);
    float* s4   = (float*)alloc((size_t)N * 4 * 4);
    float* agg  = (float*)alloc((size_t)N * 128 * 4);
    float* x0   = (float*)alloc((size_t)N * 128 * 4);
    float* hV1  = (float*)alloc((size_t)N * 128 * 4);
    float* x1   = (float*)alloc((size_t)N * 128 * 4);
    float* bb3p = (float*)alloc(16 * 4);
    float* meanB = (float*)alloc(128 * 4);
    float* invB  = (float*)alloc(128 * 4);
    (void)ws_size; (void)n_in; (void)out_size;

    auto grd = [](long long n) { return dim3((unsigned)((n + 255) / 256)); };
    const int ntE = (E + 15) / 16;
    const int ntN = (N + 15) / 16;
    dim3 gE((unsigned)((ntE + 3) / 4)), gN((unsigned)((ntN + 3) / 4)), blk(128);

    // 1) fp32 -> fp16 conversions
    k_f2h<<<grd((long long)N * 128), 256, 0, stream>>>(hV32, hV16, N * 128);
    k_f2h<<<grd((long long)E * 128), 256, 0, stream>>>(hE32, hE16, E * 128);
    k_f2h<<<grd(128 * 256), 256, 0, stream>>>(Wv1, Wv116, 128 * 256);
    k_f2h<<<grd(128 * 128), 256, 0, stream>>>(Wv2, Wv216, 128 * 128);
    k_f2h<<<grd(128 * 128), 256, 0, stream>>>(Wv3, Wv316, 128 * 128);
    k_f2h<<<grd(128 * 384), 256, 0, stream>>>(Wb1, Wb116, 128 * 384);
    k_f2h<<<grd(128 * 128), 256, 0, stream>>>(Wb2, Wb216, 128 * 128);
    k_padwb3<<<grd(16 * 128), 256, 0, stream>>>(Wb3, bb3, Wb3p, bb3p);
    k_f2h<<<grd(128 * 128), 256, 0, stream>>>(Wo, Wo16, 128 * 128);
    k_f2h<<<grd(512 * 128), 256, 0, stream>>>(Wd1, Wd116, 512 * 128);
    k_f2h<<<grd(128 * 512), 256, 0, stream>>>(Wd2, Wd216, 128 * 512);
    k_f2h<<<grd(128 * 384), 256, 0, stream>>>(We1, We116, 128 * 384);
    k_f2h<<<grd(128 * 128), 256, 0, stream>>>(We2, We216, 128 * 128);
    k_f2h<<<grd(128 * 128), 256, 0, stream>>>(We3, We316, 128 * 128);

    // 2) edge attention MLPs (WMMA) -> V16, wE
    k_edge_attn<<<gE, blk, 0, stream>>>(hE16, hV16, eidx,
        Wv116, bv1, Wv216, bv2, Wv316, bv3,
        Wb116, bb1, Wb216, bb2, Wb3p, bb3p,
        V16, wE, E, ntE);

    // 3) segment softmax aggregation (unnormalized; normalized in k_node_wo)
    k_init<<<grd((long long)N * 128), 256, 0, stream>>>(m4, s4, agg, N);
    k_segmax<<<grd((long long)E * 4), 256, 0, stream>>>(wE, eidx, m4, E);
    k_segsum<<<grd((long long)E * 4), 256, 0, stream>>>(wE, m4, eidx, s4, E);
    k_segagg<<<grd((long long)E * 128), 256, 0, stream>>>(wE, m4, eidx, V16, agg, E);

    // 4) dh = (agg/s) @ Wo^T ; x0 = hV + dh
    k_node_wo<<<gN, blk, 0, stream>>>(agg, s4, Wo16, hV32, x0, N, ntN);

    // 5) batchnorm 0 -> hV1 (fp32 + fp16)
    k_stats<<<128, 256, 0, stream>>>(x0, N, meanB, invB);
    k_norm<<<grd((long long)N * 128), 256, 0, stream>>>(x0, meanB, invB, g0, b0, hV1, hV1_16, (long long)N * 128);

    // 6) FFN
    k_ffn1<<<gN, blk, 0, stream>>>(hV1_16, Wd116, bd1, t16, N, ntN);
    k_ffn2<<<gN, blk, 0, stream>>>(t16, Wd216, bd2, hV1, x1, N, ntN);

    // 7) batchnorm 1 -> final h_V (d_out) + fp16 copy for EdgeMLP
    k_stats<<<128, 256, 0, stream>>>(x1, N, meanB, invB);
    k_norm<<<grd((long long)N * 128), 256, 0, stream>>>(x1, meanB, invB, g1, b1, out_hV, hV2_16, (long long)N * 128);

    // 8) EdgeMLP -> pre-norm edge output staged in d_out
    k_edge_mlp<<<gE, blk, 0, stream>>>(hV2_16, hE16, hE32, eidx,
        We116, be1, We216, be2, We316, be3,
        out_hE, E, ntE);

    // 9) edge batchnorm in place on d_out
    k_stats<<<128, 256, 0, stream>>>(out_hE, E, meanB, invB);
    k_norm<<<grd((long long)E * 128), 256, 0, stream>>>(out_hE, meanB, invB, ge, bEn, out_hE, (h16*)nullptr, (long long)E * 128);
}
